// DecisionTransformer_22471268893077
// MI455X (gfx1250) — matrix operand order repaired and checked by reading, not compiled, not run on hardware
//
#include <hip/hip_runtime.h>
#include <hip/hip_bf16.h>
#include <math.h>

// ---------------- problem constants ----------------
constexpr int Bb  = 2;
constexpr int Tt  = 512;
constexpr int Cc  = 512;
constexpr int Ll  = 3;
constexpr int NHh = 8;
constexpr int Dd  = 64;          // head dim
constexpr int Ss  = 3 * Tt;      // 1536
constexpr int H4  = 4 * Cc;      // 2048
constexpr int SDd = 65;
constexpr int ADd = 17;

// output offsets (floats), reference return order:
// (state_preds, action_preds, rtg_preds, all_w, all_m)
constexpr long long OFF_SP = 0;
constexpr long long OFF_AP = OFF_SP + (long long)Bb * Tt * SDd;
constexpr long long OFF_RP = OFF_AP + (long long)Bb * Tt * ADd;
constexpr long long OFF_W  = OFF_RP + (long long)Bb * Tt;
constexpr long long WSZ_L  = (long long)Bb * NHh * Ss * Ss;       // per-layer attn maps
constexpr long long OFF_M  = OFF_W + (long long)Ll * WSZ_L;
constexpr long long MSZ_L  = (long long)Bb * Ss * H4;

typedef __attribute__((ext_vector_type(16))) __bf16 v16bf;
typedef __attribute__((ext_vector_type(8)))  float  v8f;

// flags
#define FLG_MASK 1
#define FLG_GELU 2
// store modes
#define SM_NORMAL 0
#define SM_QK     1   // [B*S, C] -> [B,NH,S,D]
#define SM_VT     2   // [B*S, C] -> [B,NH,D,S]
#define SM_ATT    3   // batched per z=(b*NH+h): [S,D] -> h[(b*S+s)*C + h*D + d]

// ================= generic bf16 WMMA GEMM =================
// C = scale * (A[M,K] @ W[N,K]^T) (+bias[N]) (gelu) (+resid) (mask) -> store mode
// Block tile 128x64x64: 256 threads = 8 wave32 waves (4 M x 2 N), wave tile 32x32,
// 8 wmma/wave/iter (2 M-subtiles x 2 N-subtiles x 2 K-chunks of 32).
// Double-buffered LDS staging with fp32->bf16 conversion on the staging path
// (global_load_b128 -> v_cvt_pk_bf16_f32 -> ds_store_b128).
// Exploited: M % 128 == 0, K % 64 == 0, lda/ldb % 4 == 0 at every call site; only N
// may be ragged (heads: 65/17/1) -> row-clamped loads, guarded stores.
__global__ __launch_bounds__(256)
void gemm_bf16_kernel(const float* __restrict__ A,  long long strideA,
                      const float* __restrict__ Bw, long long strideB,
                      const float* __restrict__ bias,
                      const float* __restrict__ resid,
                      float* __restrict__ Cout, long long strideC,
                      int M, int N, int K, int lda, int ldb, int ldc,
                      float scale, int flags, int smode)
{
    constexpr int BM = 128, BN = 64, BK = 64, PAD = 8, LDSW = BK + PAD; // 72 bf16 / row
    __shared__ __attribute__((aligned(16))) __bf16 sA[2][BM][LDSW];
    __shared__ __attribute__((aligned(16))) __bf16 sB[2][BN][LDSW];

    const int z = blockIdx.z;
    A  += (long long)z * strideA;
    Bw += (long long)z * strideB;
    Cout += (long long)z * strideC;

    const int m0 = blockIdx.x * BM;
    const int n0 = blockIdx.y * BN;
    const int tid  = threadIdx.x;
    const int lane = tid & 31;
    const int wave = tid >> 5;
    const int wm   = wave & 3;      // 0..3 -> 32-row M slab
    const int wn   = wave >> 2;     // 0..1 -> 32-col N slab
    const int half = lane >> 4;     // 0/1
    const int r    = lane & 15;

    // staging ownership:
    //  A: thread -> (row 0..127, 32-float segment 0..1)  : 8 b128 loads, 4 b128 LDS stores
    //  B: thread -> (row 0..63,  16-float segment 0..3)  : 4 b128 loads, 2 b128 LDS stores
    const int sarow = tid >> 1;
    const int sacol = (tid & 1) * 32;
    const int sbrow = tid >> 2;
    const int sbcol = (tid & 3) * 16;
    const int bgn   = n0 + sbrow;
    const float* arow_p = A  + (long long)(m0 + sarow) * lda + sacol;   // m0+sarow < M
    const float* brow_p = Bw + (long long)((bgn < N) ? bgn : (N - 1)) * ldb + sbcol;

    auto pack8 = [](float4 x0, float4 x1) {
        union { __bf16 hh[8]; uint4 u; } p;
        p.hh[0] = (__bf16)x0.x; p.hh[1] = (__bf16)x0.y;
        p.hh[2] = (__bf16)x0.z; p.hh[3] = (__bf16)x0.w;
        p.hh[4] = (__bf16)x1.x; p.hh[5] = (__bf16)x1.y;
        p.hh[6] = (__bf16)x1.z; p.hh[7] = (__bf16)x1.w;
        return p.u;
    };

    auto stage = [&](int bufi, int k0) {
#pragma unroll
        for (int seg = 0; seg < 4; ++seg) {
            float4 a0 = *(const float4*)(arow_p + k0 + seg * 8);
            float4 a1 = *(const float4*)(arow_p + k0 + seg * 8 + 4);
            *(uint4*)&sA[bufi][sarow][sacol + seg * 8] = pack8(a0, a1); // ds_store_b128
        }
#pragma unroll
        for (int seg = 0; seg < 2; ++seg) {
            float4 b0 = *(const float4*)(brow_p + k0 + seg * 8);
            float4 b1 = *(const float4*)(brow_p + k0 + seg * 8 + 4);
            *(uint4*)&sB[bufi][sbrow][sbcol + seg * 8] = pack8(b0, b1);
        }
    };

    v8f acc[2][2] = {{{}, {}}, {{}, {}}};   // [M-subtile][N-subtile]

    stage(0, 0);
    int buf = 0;
    for (int k0 = 0; k0 < K; k0 += BK) {
        __syncthreads();                              // staged tile visible

        // build wave32 fragments per ISA 7.12.2 for both 32-wide K chunks
        v16bf af[2][2], bf[2][2];                     // [subtile][K-chunk]
#pragma unroll
        for (int c = 0; c < 2; ++c) {
            int kc = 32 * c;
#pragma unroll
            for (int e = 0; e < 16; ++e) {
                int p = e >> 1, w0 = e & 1;
                // A 16x32 bf16: lane-half 0 holds K 0..7 / 16..23, half 1 -> 8..15 / 24..31
                int ka = kc + ((p < 4) ? (2 * p) : (16 + 2 * (p - 4))) + 8 * half + w0;
                af[0][c][e] = sA[buf][wm * 32 + r][ka];
                af[1][c][e] = sA[buf][wm * 32 + 16 + r][ka];
                // B 32x16 bf16: lane = column, half 0 holds K 0..7 / 16..23, half 1 -> 8..15 / 24..31
                int kb = kc + ((e < 8) ? e : (8 + e)) + 8 * half;
                bf[0][c][e] = sB[buf][wn * 32 + r][kb];
                bf[1][c][e] = sB[buf][wn * 32 + 16 + r][kb];
            }
        }

        // prefetch next tile into the other buffer while the matrix pipe runs
        if (k0 + BK < K) stage(buf ^ 1, k0 + BK);

#pragma unroll
        for (int c = 0; c < 2; ++c)
#pragma unroll
            for (int ms = 0; ms < 2; ++ms)
#pragma unroll
                for (int ns = 0; ns < 2; ++ns)
                    acc[ms][ns] = __builtin_amdgcn_wmma_f32_16x16x32_bf16(
                        false, af[ms][c], false, bf[ns][c],
                        (short)0, acc[ms][ns], false, false);
        buf ^= 1;
    }

    // epilogue: C/D layout: lane = N (mod 16), VGPR i -> M = i + 8*half
#pragma unroll
    for (int ms = 0; ms < 2; ++ms) {
#pragma unroll
        for (int ns = 0; ns < 2; ++ns) {
            int gn = n0 + wn * 32 + ns * 16 + r;
            if (gn >= N) continue;
#pragma unroll
            for (int i = 0; i < 8; ++i) {
                int gm = m0 + wm * 32 + ms * 16 + 8 * half + i;   // always < M
                float val = acc[ms][ns][i] * scale;
                if (bias)              val += bias[gn];
                if (flags & FLG_GELU)  val = 0.5f * val * (1.0f + erff(val * 0.70710678118654752f));
                if (resid)             val += resid[(long long)gm * ldc + gn];
                if (flags & FLG_MASK) {
                    // block-causal: same rtg/state/action triple OR lower-triangular
                    if (!(((gm / 3) == (gn / 3)) || (gm >= gn))) val = -__builtin_inff();
                }
                if (smode == SM_NORMAL) {
                    Cout[(long long)gm * ldc + gn] = val;
                } else if (smode == SM_QK) {
                    int b = gm / Ss, s = gm - b * Ss;
                    Cout[(((long long)b * NHh + (gn >> 6)) * Ss + s) * Dd + (gn & 63)] = val;
                } else if (smode == SM_VT) {
                    int b = gm / Ss, s = gm - b * Ss;
                    Cout[(((long long)b * NHh + (gn >> 6)) * Dd + (gn & 63)) * Ss + s] = val;
                } else { // SM_ATT : z = b*NH + head, gm = s, gn = d
                    int b = z / NHh, hh = z - b * NHh;
                    Cout[((long long)b * Ss + gm) * Cc + hh * Dd + gn] = val;
                }
            }
        }
    }
}

// ================= embedding + input layernorm =================
__global__ __launch_bounds__(256)
void embed_ln_kernel(const int* __restrict__ timesteps, const int* __restrict__ states,
                     const int* __restrict__ actions,  const float* __restrict__ rtg,
                     const float* __restrict__ Wt, const float* __restrict__ Ws,
                     const float* __restrict__ Wa, const float* __restrict__ Wr_w,
                     const float* __restrict__ Wr_b, const float* __restrict__ g,
                     const float* __restrict__ bb, float* __restrict__ h)
{
    __shared__ float red[256];
    const int row = blockIdx.x;            // b*S + s
    const int b = row / Ss, s = row - b * Ss;
    const int t = s / 3, kind = s - 3 * t;
    const int ts = timesteps[b * Tt + t];
    const int tid = threadIdx.x;

    float v[2];
#pragma unroll
    for (int j = 0; j < 2; ++j) {
        int c = tid + 256 * j;
        float te = Wt[(long long)ts * Cc + c];
        float x;
        if (kind == 0)      x = rtg[b * Tt + t] * Wr_w[c] + Wr_b[c] + te;
        else if (kind == 1) x = Ws[(long long)states[b * Tt + t]  * Cc + c] + te;
        else                x = Wa[(long long)actions[b * Tt + t] * Cc + c] + te;
        v[j] = x;
    }
    // mean
    red[tid] = v[0] + v[1]; __syncthreads();
    for (int st = 128; st > 0; st >>= 1) { if (tid < st) red[tid] += red[tid + st]; __syncthreads(); }
    float mean = red[0] * (1.0f / Cc); __syncthreads();
    // var
    float d0 = v[0] - mean, d1 = v[1] - mean;
    red[tid] = d0 * d0 + d1 * d1; __syncthreads();
    for (int st = 128; st > 0; st >>= 1) { if (tid < st) red[tid] += red[tid + st]; __syncthreads(); }
    float rstd = rsqrtf(red[0] * (1.0f / Cc) + 1e-5f);

    h[(long long)row * Cc + tid]       = d0 * rstd * g[tid]       + bb[tid];
    h[(long long)row * Cc + tid + 256] = d1 * rstd * g[tid + 256] + bb[tid + 256];
}

// ================= in-place layernorm over C =================
__global__ __launch_bounds__(256)
void ln_kernel(float* __restrict__ h, const float* __restrict__ g, const float* __restrict__ bb)
{
    __shared__ float red[256];
    const int tid = threadIdx.x;
    float* x = h + (long long)blockIdx.x * Cc;
    float v0 = x[tid], v1 = x[tid + 256];
    red[tid] = v0 + v1; __syncthreads();
    for (int st = 128; st > 0; st >>= 1) { if (tid < st) red[tid] += red[tid + st]; __syncthreads(); }
    float mean = red[0] * (1.0f / Cc); __syncthreads();
    float d0 = v0 - mean, d1 = v1 - mean;
    red[tid] = d0 * d0 + d1 * d1; __syncthreads();
    for (int st = 128; st > 0; st >>= 1) { if (tid < st) red[tid] += red[tid + st]; __syncthreads(); }
    float rstd = rsqrtf(red[0] * (1.0f / Cc) + 1e-5f);
    x[tid]       = d0 * rstd * g[tid]       + bb[tid];
    x[tid + 256] = d1 * rstd * g[tid + 256] + bb[tid + 256];
}

// ================= row softmax over S (in place, on masked scores) =================
__global__ __launch_bounds__(256)
void softmax_kernel(float* __restrict__ w)
{
    __shared__ float red[256];
    const int tid = threadIdx.x;
    float* x = w + (long long)blockIdx.x * Ss;
    float v[6];
    float mx = -__builtin_inff();
#pragma unroll
    for (int j = 0; j < 6; ++j) { v[j] = x[tid + 256 * j]; mx = fmaxf(mx, v[j]); }
    red[tid] = mx; __syncthreads();
    for (int st = 128; st > 0; st >>= 1) { if (tid < st) red[tid] = fmaxf(red[tid], red[tid + st]); __syncthreads(); }
    mx = red[0]; __syncthreads();
    float sum = 0.0f;
#pragma unroll
    for (int j = 0; j < 6; ++j) { v[j] = expf(v[j] - mx); sum += v[j]; }
    red[tid] = sum; __syncthreads();
    for (int st = 128; st > 0; st >>= 1) { if (tid < st) red[tid] += red[tid + st]; __syncthreads(); }
    float inv = 1.0f / red[0];
#pragma unroll
    for (int j = 0; j < 6; ++j) x[tid + 256 * j] = v[j] * inv;
}

// ================= host-side orchestration =================
static inline void launch_gemm(hipStream_t st,
                               const float* A, long long sA, const float* Bw, long long sB,
                               const float* bias, const float* resid,
                               float* C, long long sC,
                               int M, int N, int K, int lda, int ldb, int ldc,
                               float scale, int flags, int smode, int Z)
{
    dim3 grid((M + 127) / 128, (N + 63) / 64, Z);
    gemm_bf16_kernel<<<grid, 256, 0, st>>>(A, sA, Bw, sB, bias, resid, C, sC,
                                           M, N, K, lda, ldb, ldc, scale, flags, smode);
}

extern "C" void kernel_launch(void* const* d_in, const int* in_sizes, int n_in,
                              void* d_out, int out_size, void* d_ws, size_t ws_size,
                              hipStream_t stream)
{
    const int*   timesteps = (const int*)  d_in[0];
    const int*   states    = (const int*)  d_in[1];
    const int*   actions   = (const int*)  d_in[2];
    const float* rtg       = (const float*)d_in[3];
    const float* Wt        = (const float*)d_in[4];
    const float* Wsemb     = (const float*)d_in[5];
    const float* Waemb     = (const float*)d_in[6];
    const float* Wr_w      = (const float*)d_in[7];
    const float* Wr_b      = (const float*)d_in[8];
    const float* ln_e_g    = (const float*)d_in[9];
    const float* ln_e_b    = (const float*)d_in[10];
    const float* Wq        = (const float*)d_in[11];
    const float* bq        = (const float*)d_in[12];
    const float* Wk        = (const float*)d_in[13];
    const float* bk        = (const float*)d_in[14];
    const float* Wv        = (const float*)d_in[15];
    const float* bv        = (const float*)d_in[16];
    const float* Wo        = (const float*)d_in[17];
    const float* bo        = (const float*)d_in[18];
    const float* W1        = (const float*)d_in[19];
    const float* b1        = (const float*)d_in[20];
    const float* W2        = (const float*)d_in[21];
    const float* b2        = (const float*)d_in[22];
    const float* ln1_g     = (const float*)d_in[23];
    const float* ln1_b     = (const float*)d_in[24];
    const float* ln2_g     = (const float*)d_in[25];
    const float* ln2_b     = (const float*)d_in[26];
    const float* pr_w      = (const float*)d_in[27];
    const float* pr_b      = (const float*)d_in[28];
    const float* ps_w      = (const float*)d_in[29];
    const float* ps_b      = (const float*)d_in[30];
    const float* pa_w      = (const float*)d_in[31];
    const float* pa_b      = (const float*)d_in[32];

    float* out = (float*)d_out;
    float* ws  = (float*)d_ws;

    const long long ROWS = (long long)Bb * Ss;        // 3072
    const long long ACT  = ROWS * Cc;                 // 1,572,864 floats
    float* h   = ws;
    float* q   = ws + ACT;                            // [B,NH,S,D]
    float* kb  = ws + 2 * ACT;                        // [B,NH,S,D]
    float* vt  = ws + 3 * ACT;                        // [B,NH,D,S]
    float* att = ws + 4 * ACT;                        // [B,S,C]

    // 1) embeddings + input LN -> h
    embed_ln_kernel<<<(int)ROWS, 256, 0, stream>>>(timesteps, states, actions, rtg,
                                                   Wt, Wsemb, Waemb, Wr_w, Wr_b,
                                                   ln_e_g, ln_e_b, h);

    const int ZBH = Bb * NHh;                         // 16
    const float att_scale = 1.0f / sqrtf((float)Dd); // 0.125

    for (int l = 0; l < Ll; ++l) {
        const float* Wql = Wq + (long long)l * Cc * Cc;
        const float* Wkl = Wk + (long long)l * Cc * Cc;
        const float* Wvl = Wv + (long long)l * Cc * Cc;
        const float* Wol = Wo + (long long)l * Cc * Cc;
        const float* W1l = W1 + (long long)l * H4 * Cc;
        const float* W2l = W2 + (long long)l * Cc * H4;
        float* wl = out + OFF_W + (long long)l * WSZ_L;   // [B,NH,S,S]
        float* ml = out + OFF_M + (long long)l * MSZ_L;   // [B,S,H4]

        // QKV projections (V stored transposed per head)
        launch_gemm(stream, h, 0, Wql, 0, bq + l * Cc, nullptr, q,  0,
                    (int)ROWS, Cc, Cc, Cc, Cc, Cc, 1.0f, 0, SM_QK, 1);
        launch_gemm(stream, h, 0, Wkl, 0, bk + l * Cc, nullptr, kb, 0,
                    (int)ROWS, Cc, Cc, Cc, Cc, Cc, 1.0f, 0, SM_QK, 1);
        launch_gemm(stream, h, 0, Wvl, 0, bv + l * Cc, nullptr, vt, 0,
                    (int)ROWS, Cc, Cc, Cc, Cc, Cc, 1.0f, 0, SM_VT, 1);

        // scores = scale * q @ k^T, block-causal masked -> all_w[l] (batched over b,h)
        launch_gemm(stream, q, (long long)Ss * Dd, kb, (long long)Ss * Dd,
                    nullptr, nullptr, wl, (long long)Ss * Ss,
                    Ss, Ss, Dd, Dd, Dd, Ss, att_scale, FLG_MASK, SM_NORMAL, ZBH);

        // row softmax in place
        softmax_kernel<<<ZBH * Ss, 256, 0, stream>>>(wl);

        // att = w @ v  (v supplied transposed: [D,S] per head) -> att[B,S,C]
        launch_gemm(stream, wl, (long long)Ss * Ss, vt, (long long)Dd * Ss,
                    nullptr, nullptr, att, 0,
                    Ss, Dd, Ss, Ss, Ss, Cc, 1.0f, 0, SM_ATT, ZBH);

        // h = h + att @ Wo^T + bo ; LN1
        launch_gemm(stream, att, 0, Wol, 0, bo + l * Cc, h, h, 0,
                    (int)ROWS, Cc, Cc, Cc, Cc, Cc, 1.0f, 0, SM_NORMAL, 1);
        ln_kernel<<<(int)ROWS, 256, 0, stream>>>(h, ln1_g + l * Cc, ln1_b + l * Cc);

        // m = gelu(h @ W1^T + b1) -> all_m[l]
        launch_gemm(stream, h, 0, W1l, 0, b1 + (long long)l * H4, nullptr, ml, 0,
                    (int)ROWS, H4, Cc, Cc, Cc, H4, 1.0f, FLG_GELU, SM_NORMAL, 1);

        // h = h + m @ W2^T + b2 ; LN2
        launch_gemm(stream, ml, 0, W2l, 0, b2 + l * Cc, h, h, 0,
                    (int)ROWS, Cc, H4, H4, H4, Cc, 1.0f, 0, SM_NORMAL, 1);
        ln_kernel<<<(int)ROWS, 256, 0, stream>>>(h, ln2_g + l * Cc, ln2_b + l * Cc);
    }

    // heads: hr = h.reshape(B,3,T,C); chunk 2 -> rtg/state preds, chunk 1 -> action preds
    launch_gemm(stream, h + 2LL * Tt * Cc, (long long)Ss * Cc, ps_w, 0, ps_b, nullptr,
                out + OFF_SP, (long long)Tt * SDd,
                Tt, SDd, Cc, Cc, Cc, SDd, 1.0f, 0, SM_NORMAL, Bb);
    launch_gemm(stream, h + 1LL * Tt * Cc, (long long)Ss * Cc, pa_w, 0, pa_b, nullptr,
                out + OFF_AP, (long long)Tt * ADd,
                Tt, ADd, Cc, Cc, Cc, ADd, 1.0f, 0, SM_NORMAL, Bb);
    launch_gemm(stream, h + 2LL * Tt * Cc, (long long)Ss * Cc, pr_w, 0, pr_b, nullptr,
                out + OFF_RP, (long long)Tt,
                Tt, 1, Cc, Cc, Cc, 1, 1.0f, 0, SM_NORMAL, Bb);
}
